// Network_38560216383903
// MI455X (gfx1250) — matrix operand order repaired and checked by reading, not compiled, not run on hardware
//
#include <hip/hip_runtime.h>
#include <hip/hip_bf16.h>

// Problem constants (from the reference)
constexpr int NC = 22;      // channels
constexpr int NV = 12;      // vocab
constexpr int ND = 16;      // feature dim
constexpr int NH = 8;       // hidden
constexpr int NB = 32768;   // batch

typedef __attribute__((ext_vector_type(16))) _Float16 v16h;
typedef __attribute__((ext_vector_type(8)))  _Float16 v8h;
typedef __attribute__((ext_vector_type(8)))  float    v8f;

// T table: (NC*NV + 1) rows of ND halves; last row is all-zero padding target.
constexpr int TT_ROWS   = NC * NV + 1;            // 265
constexpr int TT_HALVES = TT_ROWS * ND;           // 4240 halves = 8480 B
constexpr int TT_ZROW_B = (NC * NV) * ND * 2;     // byte offset of the zero row

// Workspace layout (bytes)
constexpr int WS_WCA = 0;          // A-operand Wc: 16d*2tile*32lane*16h halves = 32768 B
constexpr int WS_TT  = 32768;      // Ttab[c][v][d] + zero row : 8480 B
constexpr int WS_SQ  = 41248;      // sqnorm per (c,v): 264 f32 = 1056 B
constexpr int WS_CNT = 42304;      // histogram:       264 i32 = 1056 B

// ---------------------------------------------------------------------------
// Kernel A: precompute
//   Ttab[(c*NV+v)*ND+d] = b2 + sum_h tanh(emb[c,v,d]*w1[h]+b1[h])*w2[h] (f16)
//   plus one all-zero row at the end (gather target for padded lanes).
//   WcA = row-normalized W_fc pre-swizzled into this code's WMMA A-operand k order (same as B)
//         (lane m = row M; half h -> K = 16*g + h), zero-padded.
//   sqnorm[c][v] = ||emb[c,v,:]||^2 ; histogram bins zeroed.
// ---------------------------------------------------------------------------
__global__ __launch_bounds__(256) void precompute_kernel(
    const float* __restrict__ emb, const float* __restrict__ W_fc,
    const float* __restrict__ w1,  const float* __restrict__ b1,
    const float* __restrict__ w2,  const float* __restrict__ b2,
    char* __restrict__ ws)
{
  _Float16* wca = (_Float16*)(ws + WS_WCA);
  _Float16* tt  = (_Float16*)(ws + WS_TT);
  float*    sq  = (float*)(ws + WS_SQ);
  int*      cnt = (int*)(ws + WS_CNT);
  const int tid = threadIdx.x;

  const float vb2 = b2[0];
  for (int idx = tid; idx < NC * NV * ND; idx += 256) {
    const float x = emb[idx];
    float t = vb2;
    #pragma unroll
    for (int h = 0; h < NH; ++h) t += tanhf(x * w1[h] + b1[h]) * w2[h];
    tt[idx] = (_Float16)t;                  // idx == (c*NV+v)*ND + d already
  }
  if (tid < ND) tt[NC * NV * ND + tid] = (_Float16)0.f;   // zero row

  for (int p = tid; p < NC * NV; p += 256) {
    float s = 0.f;
    for (int d = 0; d < ND; ++d) { const float e = emb[p * ND + d]; s += e * e; }
    sq[p] = s;
    cnt[p] = 0;
  }

  for (int t = tid; t < 16 * 2 * 32 * 16; t += 256) {
    const int h    = t & 15;
    const int lane = (t >> 4) & 31;
    const int tile = (t >> 9) & 1;
    const int d    = t >> 10;
    const int g    = lane >> 4, m = lane & 15;
    const int i    = tile * 16 + m;                       // output row M
    const int K    = 16 * g + h;                          // A-operand K index
    float val = 0.f;
    if (i < NC && K < NC) {
      const float* row = W_fc + (i * NC + K) * ND;
      float s = 0.f;
      for (int dd = 0; dd < ND; ++dd) s += row[dd] * row[dd];
      float nrm = sqrtf(s);
      nrm = (nrm < 1.f) ? 1.f : nrm;
      val = row[d] / nrm;
    }
    wca[t] = (_Float16)val;
  }
}

// ---------------------------------------------------------------------------
// Kernel B: per-channel histogram of feats (LDS-local, merged globally).
// ---------------------------------------------------------------------------
__global__ __launch_bounds__(256) void hist_kernel(const int* __restrict__ feats,
                                                   char* __restrict__ ws)
{
  __shared__ int h[NC * NV];
  const int tid = threadIdx.x;
  if (tid < NC * NV) h[tid] = 0;
  __syncthreads();
  for (int idx = blockIdx.x * 256 + tid; idx < NC * NB; idx += gridDim.x * 256) {
    const int c = idx >> 15;                 // NB == 32768
    int v = feats[idx];
    v = (v < 0) ? 0 : ((v >= NV) ? NV - 1 : v);
    atomicAdd(&h[c * NV + v], 1);
  }
  __syncthreads();
  int* cnt = (int*)(ws + WS_CNT);
  if (tid < NC * NV) atomicAdd(&cnt[tid], h[tid]);
}

// ---------------------------------------------------------------------------
// Kernel C: regs = 1e-5 * 2C * sum_c sqrt(sum_v cnt[c,v]*sqnorm[c,v])
// ---------------------------------------------------------------------------
__global__ void regs_kernel(const char* __restrict__ ws, float* __restrict__ out)
{
  if (threadIdx.x == 0 && blockIdx.x == 0) {
    const int*   cnt = (const int*)(ws + WS_CNT);
    const float* sq  = (const float*)(ws + WS_SQ);
    float tot = 0.f;
    for (int c = 0; c < NC; ++c) {
      float s = 0.f;
      for (int v = 0; v < NV; ++v) s += (float)cnt[c * NV + v] * sq[c * NV + v];
      tot += sqrtf(s);
    }
    out[NB] = 1e-5f * (2.0f * NC) * tot;
  }
}

// ---------------------------------------------------------------------------
// Kernel D: one wave per 16-batch tile (4 waves per block).
// Gather byte-addresses are d-invariant and computed once per tile. The d
// loop runs in two halves of 8; per half, each lane loads only the 16-byte
// half-rows it needs (30 x ds_load_b128), then 8 unrolled d-iterations build
// the B operand by compile-time half extraction and issue two
// v_wmma_f32_16x16x32_f16 (U = Wc_d x T_d), finishing with the T.*U dot in
// the C/D register layout (v_fma_mix from f16 halves). sched_barrier(0)
// between the halves stops the scheduler from re-fusing them (which would
// double the live row-buffer and push allocation past 256 VGPRs).
// ---------------------------------------------------------------------------
__global__ __launch_bounds__(128) void main_kernel(const int* __restrict__ feats,
                                                   const char* __restrict__ ws,
                                                   float* __restrict__ out)
{
  __shared__ v16h     sA[16 * 2 * 32];      // 32 KB: pre-swizzled Wc A-operands
  __shared__ _Float16 sT[TT_HALVES];        // 8480 B: Ttab[c][v][d] + zero row

  const int tid = threadIdx.x;

  // Cooperative LDS fill (read-only tables, shared by the 4 waves)
  {
    const uint4* src = (const uint4*)(ws + WS_WCA);
    uint4* dst = (uint4*)sA;
    #pragma unroll
    for (int k = 0; k < 16; ++k) dst[tid + 128 * k] = src[tid + 128 * k];
    const unsigned* s2 = (const unsigned*)(ws + WS_TT);
    unsigned* d2 = (unsigned*)sT;
    for (int k = tid; k < TT_HALVES / 2; k += 128) d2[k] = s2[k];
  }
  __syncthreads();

  const int wave = tid >> 5, lane = tid & 31;
  const int g = lane >> 4, b = lane & 15;
  const int base = (blockIdx.x * 4 + wave) * 16;

  __builtin_prefetch(feats + base + b, 0, 1);   // global_prefetch path

  const char* tbytes = (const char*)sT;

  // --- Precompute all d-invariant gather byte-addresses (then f[] dies) ---
  int bbase[16];   // B operand: lane holds column b, K = 16*g + h
  int c0base[8];   // tile0 dot: i = 8g + r
  int c1base[6];   // tile1 dot: i = 16 + r (u1[r]==0 on g==1 lanes)
  {
    int f[NC];
    #pragma unroll
    for (int j = 0; j < NC; ++j) f[j] = feats[j * NB + base + b];

    #pragma unroll
    for (int h = 0; h < 16; ++h) {
      const int lo = (h * NV + f[h]) * (ND * 2);
      const int hi = (16 + h < NC) ? ((16 + h) * NV + f[16 + h]) * (ND * 2)
                                   : TT_ZROW_B;
      bbase[h] = g ? hi : lo;
    }
    #pragma unroll
    for (int r = 0; r < 8; ++r) {
      const int lo = (r * NV + f[r]) * (ND * 2);
      const int hi = ((8 + r) * NV + f[8 + r]) * (ND * 2);
      c0base[r] = g ? hi : lo;
    }
    #pragma unroll
    for (int r = 0; r < 6; ++r)
      c1base[r] = ((16 + r) * NV + f[16 + r]) * (ND * 2);
  }

  __builtin_amdgcn_sched_barrier(0);

  float y = 0.f;

  #pragma unroll
  for (int dh = 0; dh < 2; ++dh) {            // two halves of the d range
    const int doff = dh * 16;                 // byte offset of this half-row

    // Load the 16-byte half-rows this half needs (one ds_load_b128 each).
    v8h brow[16], c0row[8], c1row[6];
    #pragma unroll
    for (int h = 0; h < 16; ++h)
      brow[h] = *(const v8h*)(tbytes + bbase[h] + doff);
    #pragma unroll
    for (int r = 0; r < 8; ++r)
      c0row[r] = *(const v8h*)(tbytes + c0base[r] + doff);
    #pragma unroll
    for (int r = 0; r < 6; ++r)
      c1row[r] = *(const v8h*)(tbytes + c1base[r] + doff);

    #pragma unroll
    for (int d2 = 0; d2 < 8; ++d2) {
      const int d = dh * 8 + d2;
      const v16h a0 = sA[(d * 2 + 0) * 32 + lane];    // rows  0..15 of Wc_d
      const v16h a1 = sA[(d * 2 + 1) * 32 + lane];    // rows 16..21 (rest 0)

      v16h bm;
      #pragma unroll
      for (int h = 0; h < 16; ++h) bm[h] = brow[h][d2];

      const v8f cz = {};
      const v8f u0 = __builtin_amdgcn_wmma_f32_16x16x32_f16(
          false, a0, false, bm, (short)0, cz, false, false);
      const v8f u1 = __builtin_amdgcn_wmma_f32_16x16x32_f16(
          false, a1, false, bm, (short)0, cz, false, false);

      #pragma unroll
      for (int r = 0; r < 8; ++r) y += (float)c0row[r][d2] * u0[r];
      #pragma unroll
      for (int r = 0; r < 6; ++r) y += (float)c1row[r][d2] * u1[r];
    }

    // Keep the two halves' row buffers from being co-scheduled.
    __builtin_amdgcn_sched_barrier(0);
  }

  // Fold the two M-halves (lane l and l^16 hold the same batch column).
  y += __shfl_xor(y, 16, 32);
  if (lane < 16) out[base + b] = y;
}

// ---------------------------------------------------------------------------
extern "C" void kernel_launch(void* const* d_in, const int* in_sizes, int n_in,
                              void* d_out, int out_size, void* d_ws, size_t ws_size,
                              hipStream_t stream)
{
  const int*   feats = (const int*)  d_in[0];
  const float* emb   = (const float*)d_in[1];
  const float* W_fc  = (const float*)d_in[2];
  const float* w1    = (const float*)d_in[3];
  const float* b1    = (const float*)d_in[4];
  const float* w2    = (const float*)d_in[5];
  const float* b2    = (const float*)d_in[6];
  char*  ws  = (char*)d_ws;
  float* out = (float*)d_out;
  (void)in_sizes; (void)n_in; (void)out_size; (void)ws_size;

  hipLaunchKernelGGL(precompute_kernel, dim3(1), dim3(256), 0, stream,
                     emb, W_fc, w1, b1, w2, b2, ws);
  hipLaunchKernelGGL(hist_kernel, dim3(128), dim3(256), 0, stream, feats, ws);
  hipLaunchKernelGGL(regs_kernel, dim3(1), dim3(64), 0, stream, ws, out);
  // 2048 batch tiles of 16, 4 waves per block -> 512 blocks
  hipLaunchKernelGGL(main_kernel, dim3(512), dim3(128), 0, stream, feats, ws, out);
}
